// DeformableAttention_90486370992414
// MI455X (gfx1250) — matrix operand order repaired and checked by reading, not compiled, not run on hardware
//
#include <hip/hip_runtime.h>
#include <hip/hip_bf16.h>

// ---------------------------------------------------------------------------
// MI455X (gfx1250) deformable attention.
// Precision: bf16 WMMA (v_wmma_f32_16x16x32_bf16) with f32 accumulate.
// Roofline: ~41 GFLOP, ~200MB HBM -> memory floor ~8us @23.3TB/s; bf16 WMMA
// keeps GEMM compute at/below that floor (fp32 WMMA K=4 would be ~8x slower).
// Conv input staging uses gfx1250 async loads to LDS (ASYNCcnt) with LDS
// double buffering so the next channel-chunk streams in under the WMMAs.
// ---------------------------------------------------------------------------

typedef __attribute__((ext_vector_type(16))) __bf16 v16bf;
typedef __attribute__((ext_vector_type(8)))  float  v8f;

union FragBF {
    v16bf v;
    uint4 u[2];
};

#define NIMG 8
#define CH   256
#define HW   4096   // 64*64
#define OUTC 256
#define HEADS 8
#define K2   9
#define HD   32
#define MCONV 216   // 144 offset channels + 72 attn channels

// ---- WMMA fragment loaders (contiguous k = 16h + s for both A and B, wave32) ----------------

// A (16x32, row-major source, ldk halves between rows). Lanes 0-15 hold
// K 0..15 of row (lane&15); lanes 16-31 hold K 16..31 (contiguous).
static __device__ __forceinline__ v16bf load_a_frag(const __bf16* A, int ldk, int lane) {
    int r  = lane & 15;
    int kh = (lane & 16) ? 16 : 0;
    const __bf16* p = A + (size_t)r * ldk + kh;
    FragBF f;
    f.u[0] = *(const uint4*)(p);
    f.u[1] = *(const uint4*)(p + 8);
    return f.v;
}

// B (32x16). Source is n-major: Bt[n][k], ldk halves between n-rows.
// Lane j=lane&15 is column n=j; lanes<16 hold K 0..15, lanes>=16 K 16..31,
// contiguous (packed {2p,2p+1} per VGPR).
static __device__ __forceinline__ v16bf load_b_frag(const __bf16* Bt, int ldk, int lane) {
    int j  = lane & 15;
    int kb = (lane & 16) ? 16 : 0;
    const __bf16* p = Bt + (size_t)j * ldk + kb;
    FragBF f;
    f.u[0] = *(const uint4*)(p);
    f.u[1] = *(const uint4*)(p + 8);
    return f.v;
}

static __device__ __forceinline__ v8f wmma_bf16(v16bf a, v16bf b, v8f c) {
    return __builtin_amdgcn_wmma_f32_16x16x32_bf16(false, a, false, b, (short)0, c,
                                                   false, false);
}

static __device__ __forceinline__ v8f vzero8() {
    v8f z = {0.f, 0.f, 0.f, 0.f, 0.f, 0.f, 0.f, 0.f};
    return z;
}

// D 16x16 f32: VGPR r -> M = r + 8*(lane>=16), N = lane&15.
static __device__ __forceinline__ void store_d(float* D, int ldd, int lane, v8f c) {
    int n  = lane & 15;
    int m0 = (lane & 16) ? 8 : 0;
#pragma unroll
    for (int r = 0; r < 8; ++r) D[(size_t)(m0 + r) * ldd + n] = c[r];
}

// 64B global -> LDS async copy (4 x b128), ASYNCcnt-tracked (cdna5 08_async).
// INST_OFFSET is added to BOTH the LDS and global addresses, which is exactly
// the contiguous-copy behavior we need.
static __device__ __forceinline__ void async_copy64(unsigned lds_off,
                                                    const __bf16* g) {
    unsigned long long ga = (unsigned long long)(size_t)g;
    asm volatile(
        "global_load_async_to_lds_b128 %0, %1, off\n\t"
        "global_load_async_to_lds_b128 %0, %1, off offset:16\n\t"
        "global_load_async_to_lds_b128 %0, %1, off offset:32\n\t"
        "global_load_async_to_lds_b128 %0, %1, off offset:48"
        :: "v"(lds_off), "v"(ga)
        : "memory");
}

static __device__ __forceinline__ void wait_async0() {
    asm volatile("s_wait_asynccnt 0x0" ::: "memory");
}

// ---------------------------------------------------------------------------
// Pre-pass kernels: layout packing / precision conversion
// ---------------------------------------------------------------------------

// x [n][c][p] f32 -> xt [n][p][c] bf16 (LDS-tiled transpose, coalesced both ways)
__global__ void __launch_bounds__(256) k_xt(const float* __restrict__ X,
                                            __bf16* __restrict__ Xt) {
    __shared__ float tile[32][33];
    int n  = blockIdx.z;
    int c0 = blockIdx.y * 32;
    int p0 = blockIdx.x * 32;
    int tx = threadIdx.x & 31;
    int ty = threadIdx.x >> 5;  // 0..7
    const float* xn = X + (size_t)n * CH * HW;
#pragma unroll
    for (int s = 0; s < 32; s += 8)
        tile[ty + s][tx] = xn[(size_t)(c0 + ty + s) * HW + p0 + tx];
    __syncthreads();
    __bf16* xt = Xt + (size_t)n * HW * CH;
#pragma unroll
    for (int s = 0; s < 32; s += 8)
        xt[(size_t)(p0 + ty + s) * CH + c0 + tx] = (__bf16)tile[tx][ty + s];
}

// Wv, Wo f32 -> bf16 (row-major kept; Wv rows are the n-major B tiles of the
// v-GEMM, Wo rows are the A tiles of the output GEMM).
__global__ void __launch_bounds__(256) k_wcvt(const float* __restrict__ Wv,
                                              const float* __restrict__ Wo,
                                              __bf16* __restrict__ WvB,
                                              __bf16* __restrict__ WoB) {
    int t = blockIdx.x * 256 + threadIdx.x;
    if (t < OUTC * CH) {
        WvB[t] = (__bf16)Wv[t];
        WoB[t] = (__bf16)Wo[t];
    }
}

// Pack conv weights: Wp[tap][oc(0..255, zero-padded >=216)][c] bf16.
// off_w [144][256][3][3], attn_w [72][256][3][3]; tap = ky*3+kx.
__global__ void __launch_bounds__(256) k_wpack(const float* __restrict__ offw,
                                               const float* __restrict__ attw,
                                               __bf16* __restrict__ Wp) {
    int t = blockIdx.x * 256 + threadIdx.x;
    if (t >= 9 * 256 * CH) return;
    int tap = t >> 16;          // /65536
    int rem = t & 65535;
    int oc  = rem >> 8;
    int c   = rem & 255;
    float v = 0.f;
    if (oc < 144)       v = offw[((size_t)(oc * CH + c)) * 9 + tap];
    else if (oc < MCONV) v = attw[((size_t)((oc - 144) * CH + c)) * 9 + tap];
    Wp[t] = (__bf16)v;
}

// ---------------------------------------------------------------------------
// GEMM 1: vt[n][p][o] = sum_c xt[n][p][c] * Wv[o][c]      (M=32768,N=256,K=256)
// 8 waves per block as 4(M) x 2(N); wave tile 32x64 = 2x4 wmma tiles.
// ---------------------------------------------------------------------------
__global__ void __launch_bounds__(256) k_gemm_v(const __bf16* __restrict__ Xt,
                                                const __bf16* __restrict__ WvB,
                                                float* __restrict__ Vt) {
    int lane = threadIdx.x & 31;
    int wave = threadIdx.x >> 5;
    int wm = wave & 3;
    int wn = wave >> 2;
    int m0 = blockIdx.x * 128 + wm * 32;   // pixel (all images flattened)
    int n0 = blockIdx.y * 128 + wn * 64;   // out channel
    v8f acc[2][4];
#pragma unroll
    for (int i = 0; i < 2; ++i)
#pragma unroll
        for (int j = 0; j < 4; ++j) acc[i][j] = vzero8();

    for (int k0 = 0; k0 < CH; k0 += 32) {
        v16bf a[2], b[4];
#pragma unroll
        for (int i = 0; i < 2; ++i)
            a[i] = load_a_frag(Xt + (size_t)(m0 + i * 16) * CH + k0, CH, lane);
#pragma unroll
        for (int j = 0; j < 4; ++j)
            b[j] = load_b_frag(WvB + (size_t)(n0 + j * 16) * CH + k0, CH, lane);
#pragma unroll
        for (int i = 0; i < 2; ++i)
#pragma unroll
            for (int j = 0; j < 4; ++j) acc[i][j] = wmma_bf16(a[i], b[j], acc[i][j]);
    }
#pragma unroll
    for (int i = 0; i < 2; ++i)
#pragma unroll
        for (int j = 0; j < 4; ++j)
            store_d(Vt + (size_t)(m0 + i * 16) * OUTC + (n0 + j * 16), OUTC, lane,
                    acc[i][j]);
}

// ---------------------------------------------------------------------------
// Conv 3x3 (both convs fused, M=216 padded to 256) as 9-tap implicit GEMM.
// Per 32-channel chunk: async-copy a zero-haloed [4 rows][66 w][32 c]
// (c innermost) region into LDS with double buffering; B-frags are two
// ds_load_b128 per tile; A-frags direct from packed global weights.
// Epilogue: +bias, sigmoid for attn channels.  co[n][oc(216)][p] f32.
// ---------------------------------------------------------------------------
#define LXELEM (4 * 66 * 32)   // 8448 bf16 = 16,896 B per buffer

__global__ void __launch_bounds__(256) k_conv(const __bf16* __restrict__ Xt,
                                              const __bf16* __restrict__ Wp,
                                              const float* __restrict__ off_b,
                                              const float* __restrict__ attn_b,
                                              float* __restrict__ Co) {
    __shared__ __bf16 lx[2][LXELEM];  // 33,792 B (double buffered)
    int lane = threadIdx.x & 31;
    int wave = threadIdx.x >> 5;
    int wm = wave & 3;
    int wn = wave >> 2;
    int n  = blockIdx.z;
    int p0 = blockIdx.x * 128;            // 2 image rows
    int h0 = p0 >> 6;
    int m0 = blockIdx.y * 128 + wm * 32;  // oc base (padded space)
    const __bf16* xn = Xt + (size_t)n * HW * CH;

    v8f acc[2][4];
#pragma unroll
    for (int i = 0; i < 2; ++i)
#pragma unroll
        for (int j = 0; j < 4; ++j) acc[i][j] = vzero8();

    int lrow = threadIdx.x >> 6;  // 0..3
    int ww   = threadIdx.x & 63;
    int gr   = h0 - 1 + lrow;     // global image row for this thread's slot
    bool rok = (gr >= 0) && (gr < 64);
    const __bf16* gsrc = xn + (size_t)(gr * 64 + ww) * CH;  // + c0 per chunk
    int slot = (lrow * 66 + ww + 1) * 32;                   // bf16 elements

    // One-time zero fill: halo columns (lc = 0, 65) and out-of-image rows.
    // These LDS locations are never touched by the async copies.
    {
        uint4 z = {0u, 0u, 0u, 0u};
#pragma unroll
        for (int b = 0; b < 2; ++b) {
            if (!rok) {
                uint4* e = (uint4*)(&lx[b][slot]);
                e[0] = z; e[1] = z; e[2] = z; e[3] = z;
            }
            if (ww == 0) {
                uint4* e = (uint4*)(&lx[b][(lrow * 66 + 0) * 32]);
                e[0] = z; e[1] = z; e[2] = z; e[3] = z;
            }
            if (ww == 63) {
                uint4* e = (uint4*)(&lx[b][(lrow * 66 + 65) * 32]);
                e[0] = z; e[1] = z; e[2] = z; e[3] = z;
            }
        }
    }

    // Kick off chunk 0 (32 channels = 64B per (row,w) slot).
    if (rok) async_copy64((unsigned)(size_t)(&lx[0][slot]), gsrc + 0);

    for (int cc = 0; cc < CH / 32; ++cc) {
        // Own async writes for chunk cc complete -> barrier makes them (and the
        // one-time ds zero stores) visible to every wave in the block.
        wait_async0();
        __syncthreads();

        // Prefetch next chunk into the other buffer; it streams in under the
        // WMMAs below. Safe: all waves are past reading that buffer (barrier).
        if (cc + 1 < CH / 32 && rok)
            async_copy64((unsigned)(size_t)(&lx[(cc + 1) & 1][slot]),
                         gsrc + (cc + 1) * 32);

        const __bf16* lbuf = lx[cc & 1];
        int c0 = cc * 32;

#pragma unroll
        for (int tap = 0; tap < 9; ++tap) {
            int dy = tap / 3;
            int dx = tap % 3;
            v16bf a[2], b[4];
#pragma unroll
            for (int i = 0; i < 2; ++i)
                a[i] = load_a_frag(Wp + ((size_t)tap * 256 + m0 + i * 16) * CH + c0,
                                   CH, lane);
#pragma unroll
            for (int j = 0; j < 4; ++j) {
                int pl   = wn * 64 + j * 16 + (lane & 15);  // 0..127 local pixel
                int hrow = pl >> 6;
                int wimg = pl & 63;
                int lr   = hrow + dy;   // 0..3
                int lc   = wimg + dx;   // 0..65 (halo-padded)
                int kb   = (lane & 16) ? 16 : 0;
                const __bf16* pp = &lbuf[(lr * 66 + lc) * 32 + kb];
                FragBF f;
                f.u[0] = *(const uint4*)(pp);
                f.u[1] = *(const uint4*)(pp + 8);
                b[j] = f.v;
            }
#pragma unroll
            for (int i = 0; i < 2; ++i)
#pragma unroll
                for (int j = 0; j < 4; ++j)
                    acc[i][j] = wmma_bf16(a[i], b[j], acc[i][j]);
        }
        __syncthreads();  // all waves done with buffer cc&1 before it is refilled
    }

    // epilogue: bias + sigmoid(attn), store co[n][oc][p]
    float* con = Co + (size_t)n * MCONV * HW;
#pragma unroll
    for (int i = 0; i < 2; ++i)
#pragma unroll
        for (int j = 0; j < 4; ++j) {
            int p    = p0 + wn * 64 + j * 16 + (lane & 15);
            int mofs = (lane & 16) ? 8 : 0;
#pragma unroll
            for (int r = 0; r < 8; ++r) {
                int oc  = m0 + i * 16 + mofs + r;
                float v = acc[i][j][r];
                if (oc < 144) {
                    con[(size_t)oc * HW + p] = v + off_b[oc];
                } else if (oc < MCONV) {
                    float s = v + attn_b[oc - 144];
                    con[(size_t)oc * HW + p] = 1.0f / (1.0f + __expf(-s));
                }
            }
        }
}

// ---------------------------------------------------------------------------
// Deformable bilinear sampling + K2-weighted sum.
// wave = one (head, pixel); lane = channel (HD=32). vt is pixel-major so each
// corner gather is one coalesced 128B line. Writes combined[n][p][c] bf16
// (exactly the n-major B layout the final GEMM wants).
// ---------------------------------------------------------------------------
__global__ void __launch_bounds__(256) k_sample(const float* __restrict__ Co,
                                                const float* __restrict__ Vt,
                                                __bf16* __restrict__ Cm) {
    int lane = threadIdx.x & 31;
    int wave = threadIdx.x >> 5;
    int n    = blockIdx.z;
    int head = blockIdx.y;
    int p    = blockIdx.x * 8 + wave;
    int wi   = p & 63;
    int hi   = p >> 6;
    const float* con = Co + (size_t)n * MCONV * HW;
    const float* vn  = Vt + (size_t)n * HW * OUTC + head * HD + lane;

    float acc = 0.f;
#pragma unroll
    for (int k = 0; k < K2; ++k) {
        int cb    = head * K2 + k;
        float ox  = con[(size_t)(cb * 2 + 0) * HW + p];
        float oy  = con[(size_t)(cb * 2 + 1) * HW + p];
        float aw  = con[(size_t)(144 + cb) * HW + p];
        // xp = (gx_base + ox*2/W + 1)*0.5*(W-1) = wi + ox*63/64 (align_corners)
        float xp = (float)wi + ox * 0.984375f;
        float yp = (float)hi + oy * 0.984375f;
        float fx = floorf(xp), fy = floorf(yp);
        int x0 = (int)fx, y0 = (int)fy;
        float wx1 = xp - fx, wy1 = yp - fy;
        float wx0 = 1.f - wx1, wy0 = 1.f - wy1;
        float s = 0.f;
#pragma unroll
        for (int cy = 0; cy < 2; ++cy)
#pragma unroll
            for (int cx = 0; cx < 2; ++cx) {
                int yy = y0 + cy, xx = x0 + cx;
                float w = (cy ? wy1 : wy0) * (cx ? wx1 : wx0);
                bool valid = (yy >= 0) & (yy < 64) & (xx >= 0) & (xx < 64);
                int yc = min(max(yy, 0), 63);
                int xc = min(max(xx, 0), 63);
                float g = vn[(size_t)(yc * 64 + xc) * OUTC];
                s += valid ? w * g : 0.f;
            }
        acc += aw * s;
    }
    Cm[((size_t)n * HW + p) * OUTC + head * HD + lane] = (__bf16)acc;
}

// ---------------------------------------------------------------------------
// GEMM 2: out[n][o][p] = sum_c Wo[o][c] * combined[n][p][c] (M=256,N=32768,K=256)
// A = Wo row-major (direct frags), B = combined pixel-major (direct frags),
// coalesced f32 stores into the required [n][o][p] layout.
// ---------------------------------------------------------------------------
__global__ void __launch_bounds__(256) k_gemm_out(const __bf16* __restrict__ WoB,
                                                  const __bf16* __restrict__ Cm,
                                                  float* __restrict__ Out) {
    int lane = threadIdx.x & 31;
    int wave = threadIdx.x >> 5;
    int wm = wave & 3;
    int wn = wave >> 2;
    int m0 = blockIdx.y * 128 + wm * 32;   // out channel
    int n0 = blockIdx.x * 128 + wn * 64;   // global pixel (image-flattened)
    v8f acc[2][4];
#pragma unroll
    for (int i = 0; i < 2; ++i)
#pragma unroll
        for (int j = 0; j < 4; ++j) acc[i][j] = vzero8();

    for (int k0 = 0; k0 < CH; k0 += 32) {
        v16bf a[2], b[4];
#pragma unroll
        for (int i = 0; i < 2; ++i)
            a[i] = load_a_frag(WoB + (size_t)(m0 + i * 16) * CH + k0, CH, lane);
#pragma unroll
        for (int j = 0; j < 4; ++j)
            b[j] = load_b_frag(Cm + (size_t)(n0 + j * 16) * CH + k0, CH, lane);
#pragma unroll
        for (int i = 0; i < 2; ++i)
#pragma unroll
            for (int j = 0; j < 4; ++j) acc[i][j] = wmma_bf16(a[i], b[j], acc[i][j]);
    }

    // D tile rows = out channel, cols = pixel; out[n][o][p]
#pragma unroll
    for (int i = 0; i < 2; ++i)
#pragma unroll
        for (int j = 0; j < 4; ++j) {
            int pg   = n0 + j * 16 + (lane & 15);
            int img  = pg >> 12;            // / 4096
            int pl   = pg & 4095;
            int mofs = (lane & 16) ? 8 : 0;
#pragma unroll
            for (int r = 0; r < 8; ++r) {
                int o = m0 + i * 16 + mofs + r;
                Out[((size_t)img * OUTC + o) * HW + pl] = acc[i][j][r];
            }
        }
}

// ---------------------------------------------------------------------------
// Host launcher
// ---------------------------------------------------------------------------
extern "C" void kernel_launch(void* const* d_in, const int* in_sizes, int n_in,
                              void* d_out, int out_size, void* d_ws, size_t ws_size,
                              hipStream_t stream) {
    (void)in_sizes; (void)n_in; (void)out_size; (void)ws_size;
    const float* x      = (const float*)d_in[0];
    // d_in[1]=Wq, d_in[2]=Wk: dead code in the reference, never read.
    const float* Wv     = (const float*)d_in[3];
    const float* off_w  = (const float*)d_in[4];
    const float* off_b  = (const float*)d_in[5];
    const float* attn_w = (const float*)d_in[6];
    const float* attn_b = (const float*)d_in[7];
    const float* Wo     = (const float*)d_in[8];
    float* out = (float*)d_out;

    char* ws = (char*)d_ws;
    size_t off = 0;
    __bf16* xt  = (__bf16*)(ws + off); off += (size_t)NIMG * HW * CH * 2;      // 16 MB
    float*  vt  = (float*)(ws + off);  off += (size_t)NIMG * HW * OUTC * 4;    // 32 MB
    float*  co  = (float*)(ws + off);  off += (size_t)NIMG * MCONV * HW * 4;   // 27 MB
    __bf16* cm  = (__bf16*)(ws + off); off += (size_t)NIMG * HW * OUTC * 2;    // 16 MB
    __bf16* wvb = (__bf16*)(ws + off); off += (size_t)OUTC * CH * 2;
    __bf16* wob = (__bf16*)(ws + off); off += (size_t)OUTC * CH * 2;
    __bf16* wp  = (__bf16*)(ws + off); off += (size_t)9 * 256 * CH * 2;

    // pack / convert
    k_xt<<<dim3(HW / 32, CH / 32, NIMG), 256, 0, stream>>>(x, xt);
    k_wcvt<<<dim3((OUTC * CH + 255) / 256), 256, 0, stream>>>(Wv, Wo, wvb, wob);
    k_wpack<<<dim3((9 * 256 * CH + 255) / 256), 256, 0, stream>>>(off_w, attn_w, wp);

    // v = Wv * x  (pixel-major output for coalesced sampling gathers)
    k_gemm_v<<<dim3(NIMG * HW / 128, OUTC / 128), 256, 0, stream>>>(xt, wvb, vt);

    // fused 3x3 convs (offsets + sigmoid attention weights), async-LDS staged
    k_conv<<<dim3(HW / 128, 2, NIMG), 256, 0, stream>>>(xt, wp, off_b, attn_b, co);

    // deformable bilinear sampling -> combined (bf16, pixel-major)
    k_sample<<<dim3(HW / 8, HEADS, NIMG), 256, 0, stream>>>(co, vt, cm);

    // out = Wo * combined
    k_gemm_out<<<dim3(NIMG * HW / 128, OUTC / 128), 256, 0, stream>>>(wob, cm, out);
}
